// Conv_2259152798130
// MI455X (gfx1250) — matrix operand order repaired and checked by reading, not compiled, not run on hardware
//
#include <hip/hip_runtime.h>
#include <hip/hip_bf16.h>
#include <stdint.h>

// ---------------- types for WMMA operands ----------------
typedef __attribute__((ext_vector_type(16))) __bf16    v16bf;
typedef __attribute__((ext_vector_type(8)))  float     v8f;
typedef __attribute__((ext_vector_type(4)))  float     v4f;
typedef __attribute__((ext_vector_type(4)))  uint32_t  v4u;

union V16 {            // one bf16 WMMA A/B operand (32 bytes / lane)
  v16bf    v;
  v4u      q[2];
  uint16_t s[16];
};
union V8F {            // f32 accumulator as two 16B chunks
  v8f v;
  v4f q[2];
};

// ---------------- problem constants ----------------
#define E_EDGES   100000
#define N_NODES   10000
#define DIMF      80
#define MUL0C     32
#define MUL1C     16
#define EDGE_DIMC 32
#define HIDDENC   64
#define WNUMC     2304          // 1024 wa | 512 wc | 256 wd | 512 wb
#define NTILE     (E_EDGES / 16)   // 6250 edge tiles of 16
#define WPB       8                // waves per block (256 threads)

__device__ __forceinline__ uint16_t f2bf(float f) {
  uint32_t u = __float_as_uint(f);
  uint32_t r = u + 0x7FFFu + ((u >> 16) & 1u);   // round-to-nearest-even
  return (uint16_t)(r >> 16);
}

// ---------------- prep kernels ----------------
__global__ void zero_f32_kernel(float* __restrict__ p, int n) {
  int i = blockIdx.x * blockDim.x + threadIdx.x;
  if (i < n) p[i] = 0.0f;
}

// w2t[n][k] = bf16(mlp_w2[k][n])   (2304 x 64, row-contiguous over k)
__global__ void conv_w2_kernel(const float* __restrict__ w2, uint16_t* __restrict__ w2t) {
  int id = blockIdx.x * blockDim.x + threadIdx.x;
  if (id >= WNUMC * HIDDENC) return;
  int n = id / HIDDENC, k = id % HIDDENC;
  w2t[id] = f2bf(w2[k * WNUMC + n]);
}

// w1t[n][i] = bf16(mlp_w1[i][n])   (64 x 32)
__global__ void conv_w1_kernel(const float* __restrict__ w1, uint16_t* __restrict__ w1t) {
  int id = blockIdx.x * blockDim.x + threadIdx.x;
  if (id >= HIDDENC * EDGE_DIMC) return;
  int n = id / EDGE_DIMC, i = id % EDGE_DIMC;
  w1t[id] = f2bf(w1[i * HIDDENC + n]);
}

// ---------------- fused edge kernel ----------------
// One wave = 16 edges. GEMM2 computed transposed: D[M=wcol][N=edge],
// so per-edge contraction scales are per-lane scalars.
__global__ __launch_bounds__(256) void edge_kernel(
    const int*      __restrict__ dst,
    const float*    __restrict__ x_src,
    const float*    __restrict__ sh,
    const float*    __restrict__ edge_attr,
    const float*    __restrict__ mlp_b1,
    const float*    __restrict__ mlp_b2,
    const uint16_t* __restrict__ w2t,    // [2304][64] bf16
    const uint16_t* __restrict__ w1t,    // [64][32]  bf16
    float*          __restrict__ sums,   // [N][80]
    float*          __restrict__ cnt)    // [N]
{
  __shared__ __align__(16) uint16_t h_lds[WPB][16][64];   // hT staging, bf16
  __shared__ float x0s[WPB][16][32];                      // x0 raw
  __shared__ float x1s[WPB][16][16][3];                   // x1 raw

  const int lane = threadIdx.x & 31;
  const int wv   = threadIdx.x >> 5;
  const int tile = blockIdx.x * WPB + wv;
  if (tile >= NTILE) return;                              // per-wave exit (no barriers used)

  const int le   = lane & 15;       // this lane's edge within the tile (N index)
  const int half = lane >> 4;       // lane half selects K/M sub-ranges
  const int e    = tile * 16 + le;

  // ---- stage per-edge operands ----
  const float sh0 = sh[e * 4 + 0];
  const float s1x = sh[e * 4 + 1], s1y = sh[e * 4 + 2], s1z = sh[e * 4 + 3];
  #pragma unroll
  for (int uu = 0; uu < 16; ++uu) {
    int u = half * 16 + uu;
    x0s[wv][le][u] = x_src[e * DIMF + u];
  }
  #pragma unroll
  for (int uu = 0; uu < 8; ++uu) {
    int u = half * 8 + uu;
    #pragma unroll
    for (int m = 0; m < 3; ++m)
      x1s[wv][le][u][m] = x_src[e * DIMF + MUL0C + u * 3 + m];
  }

  // ---- GEMM1: h = relu(edge_attr @ W1 + b1), bf16 WMMA, K=32 ----
  V16 aE;   // A operand: M=edge rows, per-half K chunks {0..7,16..23} / {8..15,24..31}
  {
    const float* ea = edge_attr + e * EDGE_DIMC;
    #pragma unroll
    for (int j = 0; j < 8; ++j) aE.s[j]     = f2bf(ea[half * 8 + j]);
    #pragma unroll
    for (int j = 0; j < 8; ++j) aE.s[8 + j] = f2bf(ea[16 + half * 8 + j]);
  }
  #pragma unroll
  for (int t = 0; t < 4; ++t) {                 // 64 hidden = 4 N-tiles
    int n = t * 16 + le;
    V16 bW;                                     // B: N=hidden col, K=input dim
    const v4u* bq = (const v4u*)(w1t) + n * 4 + half;
    bW.q[0] = bq[0];
    bW.q[1] = bq[2];
    v8f hacc = {};
    hacc = __builtin_amdgcn_wmma_f32_16x16x32_bf16(false, aE.v, false, bW.v,
                                                   (short)0, hacc, false, false);
    float bias = mlp_b1[t * 16 + le];
    #pragma unroll
    for (int i = 0; i < 8; ++i) {
      float h = hacc[i] + bias;
      h = h > 0.0f ? h : 0.0f;
      h_lds[wv][half * 8 + i][t * 16 + le] = f2bf(h);   // transpose via LDS
    }
  }

  // ---- reload h as GEMM2 B operands (N=edge=lane, K=hidden), kept in regs ----
  V16 b0, b1;
  {
    const v4u* hq = (const v4u*)(&h_lds[wv][le][0]);
    b0.q[0] = hq[half];
    b0.q[1] = hq[2 + half];
    b1.q[0] = hq[4 + half];
    b1.q[1] = hq[6 + half];
  }

  // one 16-col tile of w = h @ W2 + b2, transposed: D[M=wcol][N=edge]
  auto mm_tile = [&](int c) -> v8f {
    int n = c * 16 + le;                              // W2 column (= A-matrix row)
    const v4u* wq = (const v4u*)(w2t) + n * 8 + half;
    // keep the L2-resident weight stream warm ~4 tiles ahead
    __builtin_prefetch(w2t + (size_t)(n + 64) * HIDDENC, 0, 0);
    V16 a0, a1;
    a0.q[0] = wq[0]; a0.q[1] = wq[2];                 // K 0..31   (per-half chunks)
    a1.q[0] = wq[4]; a1.q[1] = wq[6];                 // K 32..63
    V8F acc;                                          // bias init: 2x b128
    const v4f* bq = (const v4f*)(mlp_b2) + c * 4 + half * 2;
    acc.q[0] = bq[0];
    acc.q[1] = bq[1];
    v8f r = __builtin_amdgcn_wmma_f32_16x16x32_bf16(false, a0.v, false, b0.v,
                                                    (short)0, acc.v, false, false);
    r = __builtin_amdgcn_wmma_f32_16x16x32_bf16(false, a1.v, false, b1.v,
                                                (short)0, r, false, false);
    return r;
  };

  // ---- main loops: 144 column tiles of W2 (w never hits HBM) ----
  v8f o0a = {}, o0b = {};                 // out0: wout 0..15 / 16..31
  v8f o1m0 = {}, o1m1 = {}, o1m2 = {};    // out1[w][m], w = half*8+i
  const float INV_SQRT3 = 0.57735026918962576f;

  // wa: c = 0..63, u = c>>1 ; out0[w] += wa[u][w] * x0[u] * sh0
  for (int u = 0; u < 32; ++u) {
    float s = x0s[wv][le][u] * sh0;
    o0a += mm_tile(2 * u)     * s;
    o0b += mm_tile(2 * u + 1) * s;
  }
  // wc: c = 64..95 ; out1[w][m] += wc[u][w] * x0[u] * sh1[m]
  for (int u = 0; u < 32; ++u) {
    v8f acc = mm_tile(64 + u);
    float x0u = x0s[wv][le][u];
    o1m0 += acc * (x0u * s1x);
    o1m1 += acc * (x0u * s1y);
    o1m2 += acc * (x0u * s1z);
  }
  // wd: c = 96..111 ; out1[w][m] += wd[u][w] * x1[u][m] * sh0
  for (int u = 0; u < 16; ++u) {
    v8f acc = mm_tile(96 + u);
    o1m0 += acc * (x1s[wv][le][u][0] * sh0);
    o1m1 += acc * (x1s[wv][le][u][1] * sh0);
    o1m2 += acc * (x1s[wv][le][u][2] * sh0);
  }
  // wb: c = 112..143 ; out0[w] += wb[u][w] * (x1[u].sh1) / sqrt3
  for (int u = 0; u < 16; ++u) {
    float y1 = x1s[wv][le][u][0] * s1x + x1s[wv][le][u][1] * s1y
             + x1s[wv][le][u][2] * s1z;
    float s = INV_SQRT3 * y1;
    o0a += mm_tile(112 + 2 * u) * s;
    o0b += mm_tile(113 + 2 * u) * s;
  }

  // ---- scatter: segment sum via f32 atomics ----
  const float NORM = 0.14433756729740645f;   // 1/sqrt(48)
  const int d = dst[e];
  float* srow = sums + d * DIMF;
  #pragma unroll
  for (int i = 0; i < 8; ++i) {
    int w = half * 8 + i;
    atomicAdd(srow + w,       NORM * o0a[i]);
    atomicAdd(srow + 16 + w,  NORM * o0b[i]);
    atomicAdd(srow + 32 + w * 3 + 0, NORM * o1m0[i]);
    atomicAdd(srow + 32 + w * 3 + 1, NORM * o1m1[i]);
    atomicAdd(srow + 32 + w * 3 + 2, NORM * o1m2[i]);
  }
  if (lane < 16) atomicAdd(cnt + d, 1.0f);
}

// ---------------- node kernel: residual + mean ----------------
__global__ void node_kernel(const float* __restrict__ x_dst,
                            const float* __restrict__ rw0,
                            const float* __restrict__ rw1,
                            const float* __restrict__ sums,
                            const float* __restrict__ cnt,
                            float*       __restrict__ out) {
  int id = blockIdx.x * blockDim.x + threadIdx.x;
  if (id >= N_NODES * DIMF) return;
  int n = id / DIMF, col = id % DIMF;
  float res;
  if (col < MUL0C) {
    float acc = 0.0f;
    #pragma unroll
    for (int u = 0; u < MUL0C; ++u) acc += x_dst[n * DIMF + u] * rw0[u * MUL0C + col];
    res = acc * 0.17677669529663687f;            // 1/sqrt(32)
  } else {
    int cc = col - MUL0C, w = cc / 3, m = cc % 3;
    float acc = 0.0f;
    #pragma unroll
    for (int u = 0; u < MUL1C; ++u) acc += x_dst[n * DIMF + MUL0C + u * 3 + m] * rw1[u * MUL1C + w];
    res = acc * 0.25f;                           // 1/sqrt(16)
  }
  float cv = cnt[n];
  out[id] = res + sums[id] / (cv > 1.0f ? cv : 1.0f);
}

// ---------------- launcher ----------------
extern "C" void kernel_launch(void* const* d_in, const int* in_sizes, int n_in,
                              void* d_out, int out_size, void* d_ws, size_t ws_size,
                              hipStream_t stream) {
  const int*   dst       = (const int*)  d_in[0];
  const float* x_src     = (const float*)d_in[1];
  const float* x_dst     = (const float*)d_in[2];
  const float* sh        = (const float*)d_in[3];
  const float* edge_attr = (const float*)d_in[4];
  const float* mlp_w1    = (const float*)d_in[5];
  const float* mlp_b1    = (const float*)d_in[6];
  const float* mlp_w2    = (const float*)d_in[7];
  const float* mlp_b2    = (const float*)d_in[8];
  const float* res_w0    = (const float*)d_in[9];
  const float* res_w1    = (const float*)d_in[10];
  float* out = (float*)d_out;

  // workspace layout
  char* ws = (char*)d_ws;
  float*    sums = (float*)ws;                                        // N*80 f32
  float*    cnt  = (float*)(ws + (size_t)N_NODES * DIMF * 4);         // N f32
  uint16_t* w2t  = (uint16_t*)(ws + (size_t)N_NODES * DIMF * 4 + (size_t)N_NODES * 4);
  uint16_t* w1t  = w2t + (size_t)WNUMC * HIDDENC;

  // 1) zero accumulators (every call: graph replay safe)
  {
    int n = N_NODES * DIMF + N_NODES;
    zero_f32_kernel<<<(n + 255) / 256, 256, 0, stream>>>(sums, n);
  }
  // 2) weight conversion f32 -> bf16 (transposed for WMMA operand layout)
  conv_w2_kernel<<<(WNUMC * HIDDENC + 255) / 256, 256, 0, stream>>>(mlp_w2, w2t);
  conv_w1_kernel<<<(HIDDENC * EDGE_DIMC + 255) / 256, 256, 0, stream>>>(mlp_w1, w1t);
  // 3) fused edge MLP + tensor-product contraction + atomic scatter
  {
    int blocks = (NTILE + WPB - 1) / WPB;
    edge_kernel<<<blocks, 256, 0, stream>>>(dst, x_src, sh, edge_attr,
                                            mlp_b1, mlp_b2, w2t, w1t, sums, cnt);
  }
  // 4) node residual + mean
  {
    int n = N_NODES * DIMF;
    node_kernel<<<(n + 255) / 256, 256, 0, stream>>>(x_dst, res_w0, res_w1, sums, cnt, out);
  }
}
